// DynamicConv_39281770889901
// MI455X (gfx1250) — matrix operand order repaired and checked by reading, not compiled, not run on hardware
//
#include <hip/hip_runtime.h>

// ---------------------------------------------------------------------------
// DynamicConv fused kernel for gfx1250 (MI455X), wave32 + WMMA bf16 + TDM.
//   out[t,o] = sum_i x[t,i]*(e[t]·Ww[i*64+o] + bw[i*64+o]) + e[t]·Wb[o] + bb[o]
// Per 16-token tile: run the 16x4160 "gen" GEMM chunk-by-chunk with
// v_wmma_f32_16x16x32_bf16 (C seeded with b_weight), scale each 16x16 result
// tile per-row by x[t,i], accumulate into 4 persistent 16x16 f32 tiles.
// Weight chunks (16 rows x 256 cols bf16 = 8KB) are streamed into a
// double-buffered LDS staging area by the Tensor Data Mover (async), shared
// by all 8 waves of the workgroup.
// ---------------------------------------------------------------------------

typedef __attribute__((ext_vector_type(16))) __bf16       v16bf;
typedef __attribute__((ext_vector_type(8)))  float        v8f;
typedef __attribute__((ext_vector_type(4)))  unsigned int v4u;
typedef __attribute__((ext_vector_type(8)))  int          v8i;
typedef __attribute__((ext_vector_type(4)))  int          v4i;

#define C_IN        64
#define C_OUT       64
#define C_EMB       256
#define NW_ROWS     4096                 // C_IN * C_OUT
#define TOTAL_ROWS  4160                 // + C_OUT rows of W_bias
#define NCHUNK      260                  // TOTAL_ROWS / 16
#define CHUNK_HALFS (16 * C_EMB)         // one 16-row bf16 chunk = 8KB
#define TOK_PER_BLK 128
#define THREADS     256                  // 8 wave32
#define WAVES       8

__device__ __forceinline__ unsigned short f32_bf16(float f) {
    unsigned int u = __float_as_uint(f);
    u += 0x7FFFu + ((u >> 16) & 1u);     // round-to-nearest-even
    return (unsigned short)(u >> 16);
}

// Pre-pass: convert [W_weight ; W_bias] (f32, [4160][256]) -> bf16 in workspace.
__global__ void prep_bf16_kernel(const float* __restrict__ Ww,
                                 const float* __restrict__ Wb,
                                 unsigned short* __restrict__ out16) {
    int gid = blockIdx.x * blockDim.x + threadIdx.x;
    const int nw = NW_ROWS * C_EMB;
    float v = (gid < nw) ? Ww[gid] : Wb[gid - nw];
    out16[gid] = f32_bf16(v);
}

union AV { uint4 q[2]; v16bf v; };

// LDS byte offset of a shared-memory object (low 32 bits of its address).
__device__ __forceinline__ unsigned lds_off_of(const void* p) {
    return (unsigned)(unsigned long long)p;
}

// Issue one TDM load: 16x256 bf16 chunk, global (row-major, stride 256) -> LDS.
// D# built per CDNA5 ISA 8.3/8.4: group0 {count=1, lds_addr, global_addr, type=2},
// group1 {data_size=2B, tensor_dim0=256, tensor_dim1=16, tile 256x16, stride 256}.
// Groups 2/3 (and the trailing group on this toolchain) are zero: 2D tensor.
__device__ __forceinline__ void tdm_load_chunk(const unsigned short* gsrc,
                                               unsigned lds_byte_off) {
    unsigned long long ga = (unsigned long long)gsrc;
    v4u g0;
    g0[0] = 1u;                                              // count=1, user desc
    g0[1] = lds_byte_off;                                    // lds_addr (bytes)
    g0[2] = (unsigned)(ga & 0xFFFFFFFFu);                    // global_addr[31:0]
    g0[3] = (unsigned)((ga >> 32) & 0x01FFFFFFu) | (2u << 30); // addr[56:32]|type=2
    v8i g1;
    g1[0] = 0x00010000;                  // data_size code 1 = 2 bytes
    g1[1] = (int)(256u << 16);           // tensor_dim0 = 256 elements
    g1[2] = (int)(16u << 16);            // tensor_dim1 = 16 rows
    g1[3] = (int)(256u << 16);           // tile_dim0 = 256
    g1[4] = 16;                          // tile_dim1 = 16, tile_dim2 = 0
    g1[5] = 256;                         // tensor_dim0_stride = 256 elements
    g1[6] = 0;
    g1[7] = 0;
    v4i z4 = {0, 0, 0, 0};
    v8i z8 = {0, 0, 0, 0, 0, 0, 0, 0};
    __builtin_amdgcn_tensor_load_to_lds(g0, g1, z4, z4, z8, 0);
}

__global__ __launch_bounds__(THREADS)
void dynconv_wmma_kernel(const float* __restrict__ x,
                         const float* __restrict__ e,
                         const float* __restrict__ b_weight,
                         const float* __restrict__ b_bias,
                         const unsigned short* __restrict__ Wbf,  // [4160][256] bf16
                         float* __restrict__ out) {
    __shared__ __align__(16) unsigned short Ebuf[TOK_PER_BLK * C_EMB]; // 64KB bf16
    __shared__ __align__(16) float          Xbuf[TOK_PER_BLK * C_IN];  // 32KB f32
    __shared__ __align__(16) unsigned short Bbuf[2 * CHUNK_HALFS];     // 16KB dbl-buf

    const int tid  = threadIdx.x;
    const int lane = tid & 31;
    const int w    = tid >> 5;        // wave id 0..7 -> 16-token tile
    const int hi   = lane >> 4;       // half-wave select
    const int col  = lane & 15;       // N column / M row index
    const int tok0 = blockIdx.x * TOK_PER_BLK;

    // kick off async TDM load of weight chunk 0 into Bbuf[0] (wave 0 only)
    if (w == 0) tdm_load_chunk(Wbf, lds_off_of(&Bbuf[0]));

    // ---- stage E tile (f32 -> bf16) into LDS, vectorized b128 loads ----
    {
        const float4* e4 = (const float4*)(e + (size_t)tok0 * C_EMB);
        ushort4* d4 = (ushort4*)Ebuf;
        #pragma unroll
        for (int j = 0; j < (TOK_PER_BLK * C_EMB) / (THREADS * 4); ++j) {
            int idx = tid + j * THREADS;
            float4 v = e4[idx];
            ushort4 h;
            h.x = f32_bf16(v.x); h.y = f32_bf16(v.y);
            h.z = f32_bf16(v.z); h.w = f32_bf16(v.w);
            d4[idx] = h;
        }
    }
    // ---- stage X tile into LDS ----
    {
        const float4* x4 = (const float4*)(x + (size_t)tok0 * C_IN);
        float4* d4 = (float4*)Xbuf;
        #pragma unroll
        for (int j = 0; j < (TOK_PER_BLK * C_IN) / (THREADS * 4); ++j) {
            int idx = tid + j * THREADS;
            d4[idx] = x4[idx];
        }
    }
    if (w == 0) __builtin_amdgcn_s_wait_tensorcnt(0);
    __syncthreads();

    // ---- A fragments (16x32 bf16 per K-step), held in VGPRs for whole loop.
    // 16-bit A layout: lane row M = col; halfs 0..15 -> K = kk*32+hi*16+{0..15}.
    // Two ds_load_b128 per fragment.
    AV afrag[8];
    #pragma unroll
    for (int kk = 0; kk < 8; ++kk) {
        const uint4* p = (const uint4*)&Ebuf[(w * 16 + col) * C_EMB + kk * 32 + hi * 16];
        afrag[kk].q[0] = p[0];       // halfs +0..7
        afrag[kk].q[1] = p[1];       // halfs +8..15
    }

    v8f acc[4];
    #pragma unroll
    for (int t = 0; t < 4; ++t)
        #pragma unroll
        for (int r = 0; r < 8; ++r) acc[t][r] = 0.0f;

    // =================== main loop over weight chunks ===================
    #pragma unroll 1
    for (int i = 0; i < C_IN; ++i) {
        // per-row x values for this input channel (rows M = r + 8*hi)
        float xv[8];
        #pragma unroll
        for (int r = 0; r < 8; ++r)
            xv[r] = Xbuf[(w * 16 + r + 8 * hi) * C_IN + i];

        #pragma unroll
        for (int osub = 0; osub < 4; ++osub) {
            const int it = i * 4 + osub;                 // chunk id; rows n0 = it*16
            const unsigned short* Bc = Bbuf + (it & 1) * CHUNK_HALFS;

            // async-prefetch next chunk into the other LDS buffer via TDM
            if (w == 0 && it + 1 < NCHUNK)
                tdm_load_chunk(Wbf + (size_t)(it + 1) * CHUNK_HALFS,
                               lds_off_of(&Bbuf[((it + 1) & 1) * CHUNK_HALFS]));

            // seed C with b_weight[n0 + col] (constant per column)
            float bw = b_weight[it * 16 + col];
            v8f c = {bw, bw, bw, bw, bw, bw, bw, bw};

            // 8 K-steps over C_EMB=256.  B layout (32x16 bf16): lane column
            // N = col, 16 consecutive K values starting at kk*32 + hi*16.
            #pragma unroll
            for (int kk = 0; kk < 8; ++kk) {
                AV bf;
                const uint4* p = (const uint4*)&Bc[col * C_EMB + kk * 32 + hi * 16];
                bf.q[0] = p[0];
                bf.q[1] = p[1];
                c = __builtin_amdgcn_wmma_f32_16x16x32_bf16(
                        false, afrag[kk].v, false, bf.v, (short)0, c, false, false);
            }

            // out += x[t,i] * (gen tile)   (per-row scale, rows M = r + 8*hi)
            #pragma unroll
            for (int r = 0; r < 8; ++r)
                acc[osub][r] += xv[r] * c[r];

            // fence the async chunk, then release/acquire the buffers
            if (w == 0 && it + 1 < NCHUNK) __builtin_amdgcn_s_wait_tensorcnt(0);
            __syncthreads();
        }
    }

    // ======== bias chunks (256..259): out += e @ W_bias^T + b_bias ========
    #pragma unroll
    for (int osub = 0; osub < 4; ++osub) {
        const int it = NW_ROWS / 16 + osub;
        const unsigned short* Bc = Bbuf + (it & 1) * CHUNK_HALFS;

        if (w == 0 && it + 1 < NCHUNK)
            tdm_load_chunk(Wbf + (size_t)(it + 1) * CHUNK_HALFS,
                           lds_off_of(&Bbuf[((it + 1) & 1) * CHUNK_HALFS]));

        float bb = b_bias[osub * 16 + col];
        v8f c = {bb, bb, bb, bb, bb, bb, bb, bb};
        #pragma unroll
        for (int kk = 0; kk < 8; ++kk) {
            AV bf;
            const uint4* p = (const uint4*)&Bc[col * C_EMB + kk * 32 + hi * 16];
            bf.q[0] = p[0];
            bf.q[1] = p[1];
            c = __builtin_amdgcn_wmma_f32_16x16x32_bf16(
                    false, afrag[kk].v, false, bf.v, (short)0, c, false, false);
        }
        #pragma unroll
        for (int r = 0; r < 8; ++r) acc[osub][r] += c[r];

        if (it + 1 < NCHUNK) {
            if (w == 0) __builtin_amdgcn_s_wait_tensorcnt(0);
            __syncthreads();
        }
    }

    // =================== store: D tile layout M = r + 8*hi, N = col ===================
    #pragma unroll
    for (int osub = 0; osub < 4; ++osub) {
        #pragma unroll
        for (int r = 0; r < 8; ++r) {
            int t = tok0 + w * 16 + r + 8 * hi;
            out[(size_t)t * C_OUT + osub * 16 + col] = acc[osub][r];
        }
    }
}

extern "C" void kernel_launch(void* const* d_in, const int* in_sizes, int n_in,
                              void* d_out, int out_size, void* d_ws, size_t ws_size,
                              hipStream_t stream) {
    const float* x  = (const float*)d_in[0];   // [8,4096,64]
    const float* e  = (const float*)d_in[1];   // [8,4096,256]
    const float* Ww = (const float*)d_in[2];   // [4096,256]
    const float* bw = (const float*)d_in[3];   // [4096]
    const float* Wb = (const float*)d_in[4];   // [64,256]
    const float* bb = (const float*)d_in[5];   // [64]
    float* out = (float*)d_out;                // [8,4096,64]

    unsigned short* ws16 = (unsigned short*)d_ws;   // [4160][256] bf16 (~2.1 MB)

    const int total16 = TOTAL_ROWS * C_EMB;         // 1064960, multiple of 256
    prep_bf16_kernel<<<total16 / 256, 256, 0, stream>>>(Ww, Wb, ws16);

    const int ntok = 8 * 4096;
    dynconv_wmma_kernel<<<ntok / TOK_PER_BLK, THREADS, 0, stream>>>(
        x, e, bw, bb, ws16, out);
}
